// Pct_80272938762431
// MI455X (gfx1250) — compile-verified, matrix operand order checked by reading, not run on hardware
//
#include <hip/hip_runtime.h>

typedef __attribute__((ext_vector_type(16))) _Float16 v16h;
typedef __attribute__((ext_vector_type(4)))  _Float16 v4h;
typedef __attribute__((ext_vector_type(8)))  float    v8f;
typedef __attribute__((ext_vector_type(4)))  float    v4f;

#define NBATCH 32
#define NTOK   512
#define PTOK   513
#define PTA    640     // padded token rows (multiple of BM); over-read safe
#define PPAD   544     // padded att row length (multiple of BK, 16B aligned)
#define DMODEL 512
#define NHEAD  8
#define DKH    32
#define DVH    64
#define DFFN   1024

#define BM 128
#define BN 64
#define BK 32
#define LDSP 40   // padded LDS row stride in halves (8B alignment for col%4==0)

__device__ __forceinline__ float gelu_exact(float x) {
    return 0.5f * x * (1.f + erff(x * 0.70710678118654752f));
}

// ---------------------------------------------------------------------------
// Generic WMMA GEMM:  C[m,n] = epi( alpha * sum_k A[m,k] * B(n,k) )
// BKM==0: B(n,k) = Bw[n*ldb + k]   (weight [N,K] row-major)
// BKM==1: B(n,k) = Bw[k*ldb + n]   (K-major, e.g. att @ V)
// grid.z batching: off = (z/zdiv)*s1 + (z%zdiv)*s2 for A, B, C, res.
// epi: 0 none | 1 bn(scale,bias)+relu | 2 bias+gelu | 3 (+bias?)(+res?)
// REQUIREMENTS (caller guarantees): K % 32 == 0; A readable for
// ceil(M/128)*128 rows; B readable for ceil(N/64)*64 rows (BKM=0) or
// K rows x ceil(N/64)*64 cols (BKM=1); out-of-range contents finite.
// Block: 256 threads = 8 waves (4x2); wave = 32x32 tile = 4 WMMA / K-step.
// ---------------------------------------------------------------------------
template <int BKM>
__global__ __launch_bounds__(256) void gemm_wmma(
    const float* __restrict__ A, const float* __restrict__ Bw,
    float* __restrict__ C,
    int M, int N, int K, int lda, int ldb, int ldc,
    long long a_s1, long long a_s2, long long b_s1, long long b_s2,
    long long c_s1, long long c_s2, int zdiv,
    float alpha, int epi,
    const float* __restrict__ scale, const float* __restrict__ bias,
    const float* __restrict__ res, long long r_s1, long long r_s2, int ldres)
{
    __shared__ _Float16 As[BM][LDSP];
    __shared__ _Float16 Bs[BN][LDSP];

    const int z  = blockIdx.z;
    const int z0 = z / zdiv, z1 = z % zdiv;
    const float* Ap = A  + (size_t)z0 * a_s1 + (size_t)z1 * a_s2;
    const float* Bp = Bw + (size_t)z0 * b_s1 + (size_t)z1 * b_s2;
    float*       Cp = C  + (size_t)z0 * c_s1 + (size_t)z1 * c_s2;
    const float* Rp = res ? res + (size_t)z0 * r_s1 + (size_t)z1 * r_s2 : nullptr;

    const int bm = blockIdx.y * BM, bn = blockIdx.x * BN;
    const int tid  = threadIdx.x;
    const int lane = tid & 31, wave = tid >> 5;
    const int m0 = (wave >> 1) * 32;       // wave's A tiles at m0, m0+16
    const int n0 = (wave & 1) * 32;        // wave's B tiles at n0, n0+16
    const int r  = lane & 15, g = lane >> 4;
    const int g8 = g * 8;

    v8f acc00 = {}; v8f acc01 = {}; v8f acc10 = {}; v8f acc11 = {};

    for (int k0 = 0; k0 < K; k0 += BK) {
        __syncthreads();

        // ---- stage A tile (BM x BK), fp32 -> f16, unconditionally vectorized
        #pragma unroll
        for (int e = 0; e < (BM * BK) / 1024; ++e) {
            int i = (e * 256 + tid) * 4;
            int row = i >> 5, col = i & 31;
            v4f va = *(const v4f*)(Ap + (size_t)(bm + row) * lda + k0 + col);
            v4h h; h[0] = (_Float16)va[0]; h[1] = (_Float16)va[1];
                   h[2] = (_Float16)va[2]; h[3] = (_Float16)va[3];
            *(v4h*)&As[row][col] = h;
        }

        // ---- stage B tile (BN x BK), fp32 -> f16
        if (BKM) {
            #pragma unroll
            for (int e = 0; e < (BN * BK) / 1024; ++e) {
                int q = e * 256 + tid;            // quad over n, k = q>>4
                int nq = (q & 15) * 4, kk = q >> 4;
                v4f vb = *(const v4f*)(Bp + (size_t)(k0 + kk) * ldb + bn + nq);
                Bs[nq + 0][kk] = (_Float16)vb[0];
                Bs[nq + 1][kk] = (_Float16)vb[1];
                Bs[nq + 2][kk] = (_Float16)vb[2];
                Bs[nq + 3][kk] = (_Float16)vb[3];
            }
        } else {
            #pragma unroll
            for (int e = 0; e < (BN * BK) / 1024; ++e) {
                int i = (e * 256 + tid) * 4;
                int row = i >> 5, col = i & 31;
                v4f vb = *(const v4f*)(Bp + (size_t)(bn + row) * ldb + k0 + col);
                v4h h; h[0] = (_Float16)vb[0]; h[1] = (_Float16)vb[1];
                       h[2] = (_Float16)vb[2]; h[3] = (_Float16)vb[3];
                *(v4h*)&Bs[row][col] = h;
            }
        }
        __syncthreads();

        // ---- fragments per ISA 7.12.2 (lane -> row/col = lane&15, K-grp = lane>>4)
        v16h af0, af1, bf0, bf1;
        #pragma unroll
        for (int t = 0; t < 8; ++t) {
            af0[t]   = As[m0 + r][g8 + t];
            af0[8+t] = As[m0 + r][16 + g8 + t];
            af1[t]   = As[m0 + 16 + r][g8 + t];
            af1[8+t] = As[m0 + 16 + r][16 + g8 + t];
            bf0[t]   = Bs[n0 + r][g8 + t];
            bf0[8+t] = Bs[n0 + r][16 + g8 + t];
            bf1[t]   = Bs[n0 + 16 + r][g8 + t];
            bf1[8+t] = Bs[n0 + 16 + r][16 + g8 + t];
        }
        acc00 = __builtin_amdgcn_wmma_f32_16x16x32_f16(false, af0, false, bf0, (short)0, acc00, false, false);
        acc01 = __builtin_amdgcn_wmma_f32_16x16x32_f16(false, af0, false, bf1, (short)0, acc01, false, false);
        acc10 = __builtin_amdgcn_wmma_f32_16x16x32_f16(false, af1, false, bf0, (short)0, acc10, false, false);
        acc11 = __builtin_amdgcn_wmma_f32_16x16x32_f16(false, af1, false, bf1, (short)0, acc11, false, false);
    }

    // ---- store D: lane -> col = lane&15, row = (lane>>4)*8 + j
    const float bnrs = rsqrtf(1.f + 1e-5f);   // BN eval: x / sqrt(1+eps)
    v8f accs[2][2];
    accs[0][0] = acc00; accs[0][1] = acc01; accs[1][0] = acc10; accs[1][1] = acc11;
    #pragma unroll
    for (int tm = 0; tm < 2; ++tm) {
        #pragma unroll
        for (int tn = 0; tn < 2; ++tn) {
            int col = bn + n0 + tn * 16 + r;
            #pragma unroll
            for (int j = 0; j < 8; ++j) {
                int row = bm + m0 + tm * 16 + g8 + j;
                if (row < M && col < N) {
                    float v = accs[tm][tn][j] * alpha;
                    if (epi == 1) {
                        v = fmaxf(v * (scale[col] * bnrs) + bias[col], 0.f);
                    } else if (epi == 2) {
                        v = gelu_exact(v + bias[col]);
                    } else if (epi == 3) {
                        if (bias) v += bias[col];
                        if (Rp)   v += Rp[(size_t)row * ldres + col];
                    }
                    Cp[(size_t)row * ldc + col] = v;
                }
            }
        }
    }
}

// ---------------------------------------------------------------------------
// LayerNorm over last dim D (one block per row)
// ---------------------------------------------------------------------------
__global__ __launch_bounds__(256) void ln_kernel(
    const float* __restrict__ in, float* __restrict__ out,
    const float* __restrict__ g, const float* __restrict__ b,
    int rows_pb, long long in_bs, long long out_bs, int Dd)
{
    __shared__ float red[256];
    const int row = blockIdx.x;
    const int bb = row / rows_pb, m = row % rows_pb;
    const float* x = in  + (size_t)bb * in_bs  + (size_t)m * Dd;
    float*       y = out + (size_t)bb * out_bs + (size_t)m * Dd;
    const int tid = threadIdx.x;

    float s = 0.f;
    for (int i = tid; i < Dd; i += 256) s += x[i];
    red[tid] = s; __syncthreads();
    for (int o = 128; o > 0; o >>= 1) { if (tid < o) red[tid] += red[tid + o]; __syncthreads(); }
    const float mean = red[0] / (float)Dd;
    __syncthreads();

    float v = 0.f;
    for (int i = tid; i < Dd; i += 256) { float d = x[i] - mean; v += d * d; }
    red[tid] = v; __syncthreads();
    for (int o = 128; o > 0; o >>= 1) { if (tid < o) red[tid] += red[tid + o]; __syncthreads(); }
    const float inv = rsqrtf(red[0] / (float)Dd + 1e-5f);

    for (int i = tid; i < Dd; i += 256)
        y[i] = (x[i] - mean) * inv * g[i] + b[i];
}

// ---------------------------------------------------------------------------
// Policy-masked softmax (in place, rows padded to PPAD); zeroes pad columns
// so the att@V GEMM can run a tail-free K loop over PPAD.
// ---------------------------------------------------------------------------
__global__ __launch_bounds__(256) void softmax_policy_kernel(
    float* __restrict__ att, const float* __restrict__ pol)
{
    __shared__ float red[256];
    const int rid = blockIdx.x;            // z*P + l, z = b*H + h
    const int z = rid / PTOK, l = rid % PTOK;
    const int b = z / NHEAD;
    float* row = att + (size_t)z * PTA * PPAD + (size_t)l * PPAD;
    const float* pb = pol + (size_t)b * PTOK;
    const int tid = threadIdx.x;

    float m = -3.4e38f;
    for (int t = tid; t < PTOK; t += 256) m = fmaxf(m, row[t]);
    red[tid] = m; __syncthreads();
    for (int o = 128; o > 0; o >>= 1) { if (tid < o) red[tid] = fmaxf(red[tid], red[tid + o]); __syncthreads(); }
    m = red[0]; __syncthreads();

    float s = 0.f;
    for (int t = tid; t < PTOK; t += 256) {
        float mask = (t == l) ? 1.f : pb[t];
        s += expf(row[t] - m) * mask;
    }
    red[tid] = s; __syncthreads();
    for (int o = 128; o > 0; o >>= 1) { if (tid < o) red[tid] += red[tid + o]; __syncthreads(); }
    s = red[0];

    const float inv  = 1.f / (s + 1e-6f);
    const float addc = 1e-6f / (float)PTOK;
    for (int t = tid; t < PTOK; t += 256) {
        float mask = (t == l) ? 1.f : pb[t];
        row[t] = (expf(row[t] - m) * mask + addc) * inv;
    }
    for (int t = PTOK + tid; t < PPAD; t += 256) row[t] = 0.f;   // zero K-pad
}

// --------------------------- small helper kernels ---------------------------
__global__ void transpose_kernel(const float* __restrict__ x, float* __restrict__ xT) {
    int idx = blockIdx.x * 256 + threadIdx.x;            // b*512*256 + n*256 + c
    if (idx >= NBATCH * NTOK * 256) return;
    int c = idx & 255, n = (idx >> 8) & 511, b = idx >> 17;
    xT[idx] = x[((size_t)b * 256 + c) * NTOK + n];
}

__global__ void cls_kernel(float* __restrict__ H, const float* __restrict__ cls) {
    int idx = blockIdx.x * 256 + threadIdx.x;            // b*512 + d
    if (idx >= NBATCH * DMODEL) return;
    int b = idx >> 9, d = idx & 511;
    H[(size_t)b * PTA * DMODEL + (size_t)NTOK * DMODEL + d] = cls[d];
}

__global__ void prev_init_kernel(float* __restrict__ prev) {
    int idx = blockIdx.x * 256 + threadIdx.x;
    if (idx < NBATCH * NTOK) prev[idx] = 1.f;
}

__global__ __launch_bounds__(256) void prevsum_kernel(
    const float* __restrict__ prev, float* __restrict__ psum)
{
    __shared__ float red[256];
    int b = blockIdx.x, tid = threadIdx.x;
    float s = 0.f;
    for (int n = tid; n < NTOK; n += 256) s += prev[(size_t)b * NTOK + n];
    red[tid] = s; __syncthreads();
    for (int o = 128; o > 0; o >>= 1) { if (tid < o) red[tid] += red[tid + o]; __syncthreads(); }
    if (tid == 0) psum[b] = red[0];
}

__global__ void glob_kernel(const float* __restrict__ z, const float* __restrict__ prev,
                            const float* __restrict__ psum, float* __restrict__ glob)
{
    int idx = blockIdx.x * 256 + threadIdx.x;            // b*256 + c
    if (idx >= NBATCH * 256) return;
    int b = idx >> 8, c = idx & 255;
    float s = 0.f;
    for (int n = 0; n < NTOK; ++n)
        s += z[((size_t)b * NTOK + n) * DMODEL + 256 + c] * prev[(size_t)b * NTOK + n];
    glob[idx] = s / (psum[b] + 1e-6f);
}

__global__ void z2build_kernel(const float* __restrict__ z, const float* __restrict__ glob,
                               float* __restrict__ z2)
{
    int idx = blockIdx.x * 256 + threadIdx.x;            // (b*512+n)*512 + c
    if (idx >= NBATCH * NTOK * DMODEL) return;
    int c = idx & 511, b = idx >> 18;
    z2[idx] = (c < 256) ? z[idx] : glob[(size_t)b * 256 + (c - 256)];
}

__global__ void decision_kernel(const float* __restrict__ zz2,
                                const float* __restrict__ pw3, const float* __restrict__ pb3,
                                float* __restrict__ prev, float* __restrict__ pol)
{
    int idx = blockIdx.x * 256 + threadIdx.x;            // b*512 + n
    if (idx >= NBATCH * NTOK) return;
    int b = idx >> 9, n = idx & 511;
    const float* zr = zz2 + (size_t)idx * 128;
    float l0 = pb3[0], l1 = pb3[1];
    for (int j = 0; j < 128; ++j) { float zv = zr[j]; l0 += zv * pw3[j]; l1 += zv * pw3[128 + j]; }
    float m = fmaxf(l0, l1);
    float e0 = expf(l0 - m), e1 = expf(l1 - m);
    float s = e0 + e1;
    float lp0 = logf(e0 / s + 1e-8f), lp1 = logf(e1 / s + 1e-8f);
    float hard = (lp1 > lp0) ? 1.f : 0.f;                // argmax (ties -> 0)
    float hkd = hard * prev[idx];
    prev[idx] = hkd;
    pol[(size_t)b * PTOK + 1 + n] = hkd;
    if (n == 0) pol[(size_t)b * PTOK] = 1.f;
}

// --------------------------- host-side launcher -----------------------------
static inline void launch_gemm(hipStream_t st, const float* A, const float* Bw, float* C,
    int M, int N, int K, int lda, int ldb, int ldc,
    long long as1, long long as2, long long bs1, long long bs2,
    long long cs1, long long cs2, int zdiv, int zcount,
    int bkm, float alpha, int epi,
    const float* scale, const float* bias,
    const float* res, long long rs1, long long rs2, int ldres)
{
    dim3 grid((N + BN - 1) / BN, (M + BM - 1) / BM, zcount);
    if (bkm)
        gemm_wmma<1><<<grid, 256, 0, st>>>(A, Bw, C, M, N, K, lda, ldb, ldc,
            as1, as2, bs1, bs2, cs1, cs2, zdiv, alpha, epi, scale, bias, res, rs1, rs2, ldres);
    else
        gemm_wmma<0><<<grid, 256, 0, st>>>(A, Bw, C, M, N, K, lda, ldb, ldc,
            as1, as2, bs1, bs2, cs1, cs2, zdiv, alpha, epi, scale, bias, res, rs1, rs2, ldres);
}

extern "C" void kernel_launch(void* const* d_in, const int* in_sizes, int n_in,
                              void* d_out, int out_size, void* d_ws, size_t ws_size,
                              hipStream_t stream)
{
    (void)in_sizes; (void)n_in; (void)out_size; (void)ws_size;
    const float* x     = (const float*)d_in[0];
    const float* wc1   = (const float*)d_in[1];
    const float* bn1_g = (const float*)d_in[2];
    const float* bn1_b = (const float*)d_in[3];
    const float* wc2   = (const float*)d_in[4];
    const float* bn2_g = (const float*)d_in[5];
    const float* bn2_b = (const float*)d_in[6];
    const float* cls   = (const float*)d_in[7];
    const float* ln1_g = (const float*)d_in[8];
    const float* ln1_b = (const float*)d_in[9];
    const float* wq    = (const float*)d_in[10];
    const float* wk    = (const float*)d_in[11];
    const float* wv    = (const float*)d_in[12];
    const float* wfc   = (const float*)d_in[13];
    const float* ln2_g = (const float*)d_in[14];
    const float* ln2_b = (const float*)d_in[15];
    const float* ffw1  = (const float*)d_in[16];
    const float* ffb1  = (const float*)d_in[17];
    const float* ffw2  = (const float*)d_in[18];
    const float* ffb2  = (const float*)d_in[19];
    const float* pln_g = (const float*)d_in[20];
    const float* pln_b = (const float*)d_in[21];
    const float* pin_w = (const float*)d_in[22];
    const float* pin_b = (const float*)d_in[23];
    const float* pw1   = (const float*)d_in[24];
    const float* pb1   = (const float*)d_in[25];
    const float* pw2   = (const float*)d_in[26];
    const float* pb2   = (const float*)d_in[27];
    const float* pw3   = (const float*)d_in[28];
    const float* pb3   = (const float*)d_in[29];

    float* ws = (float*)d_ws;
    size_t off = 0;
    auto alloc = [&](size_t n) { float* p = ws + off; off += n; return p; };
    // token-dim buffers allocated with PTA=640 rows so 128-row tiles over-read safely
    float* xT    = alloc((size_t)NBATCH * NTOK * 256);
    float* Hb    = alloc((size_t)NBATCH * PTA * DMODEL);
    float* tmpA  = alloc((size_t)NBATCH * PTA * DMODEL);
    float* ff    = alloc((size_t)NBATCH * PTA * DFFN);
    float* qb    = alloc((size_t)NBATCH * PTA * 256);
    float* kb    = alloc((size_t)NBATCH * PTA * 256);
    float* vb    = alloc((size_t)NBATCH * PTA * DMODEL);
    float* ob    = alloc((size_t)NBATCH * PTA * DMODEL);
    float* attb  = alloc((size_t)NBATCH * NHEAD * PTA * PPAD);
    float* zb    = alloc((size_t)NBATCH * NTOK * DMODEL);
    float* z2b   = alloc((size_t)NBATCH * NTOK * DMODEL);
    float* zz1   = alloc((size_t)NBATCH * NTOK * 256);
    float* zz2   = alloc((size_t)NBATCH * NTOK * 128);
    float* globb = alloc((size_t)NBATCH * 256);
    float* psum  = alloc(NBATCH);
    float* prevb = alloc((size_t)NBATCH * NTOK);
    float* polb  = alloc((size_t)NBATCH * PTOK + 64);

    const long long HS = (long long)PTA * DMODEL;        // batch stride of H
    const long long QS = (long long)PTA * 256;           // batch stride of q/k
    const long long AS = (long long)PTA * PPAD;          // per-head att stride
    const long long FS = (long long)PTA * DFFN;

    // ---- stem: h = relu(bn2(relu(bn1(x^T @ wc1^T)) @ wc2^T)), then cls row
    transpose_kernel<<<(NBATCH * NTOK * 256 + 255) / 256, 256, 0, stream>>>(x, xT);
    launch_gemm(stream, xT, wc1, zb, NTOK, DMODEL, 256, 256, 256, DMODEL,
                (long long)NTOK * 256, 0, 0, 0, (long long)NTOK * DMODEL, 0, 1, NBATCH,
                0, 1.f, 1, bn1_g, bn1_b, nullptr, 0, 0, 0);
    launch_gemm(stream, zb, wc2, Hb, NTOK, DMODEL, DMODEL, DMODEL, DMODEL, DMODEL,
                (long long)NTOK * DMODEL, 0, 0, 0, HS, 0, 1, NBATCH,
                0, 1.f, 1, bn2_g, bn2_b, nullptr, 0, 0, 0);
    cls_kernel<<<(NBATCH * DMODEL + 255) / 256, 256, 0, stream>>>(Hb, cls);
    prev_init_kernel<<<(NBATCH * NTOK + 255) / 256, 256, 0, stream>>>(prevb);

    for (int i = 0; i < 4; ++i) {
        // ---- predictor on pts = h[:, :512]
        ln_kernel<<<NBATCH * NTOK, 256, 0, stream>>>(Hb, tmpA,
            pln_g + i * DMODEL, pln_b + i * DMODEL, NTOK, HS, HS, DMODEL);
        launch_gemm(stream, tmpA, pin_w + (size_t)i * DMODEL * DMODEL, zb,
                    NTOK, DMODEL, DMODEL, DMODEL, DMODEL, DMODEL,
                    HS, 0, 0, 0, (long long)NTOK * DMODEL, 0, 1, NBATCH,
                    0, 1.f, 2, nullptr, pin_b + i * DMODEL, nullptr, 0, 0, 0);
        prevsum_kernel<<<NBATCH, 256, 0, stream>>>(prevb, psum);
        glob_kernel<<<(NBATCH * 256 + 255) / 256, 256, 0, stream>>>(zb, prevb, psum, globb);
        z2build_kernel<<<(NBATCH * NTOK * DMODEL + 255) / 256, 256, 0, stream>>>(zb, globb, z2b);
        launch_gemm(stream, z2b, pw1 + (size_t)i * 256 * DMODEL, zz1,
                    NTOK, 256, DMODEL, DMODEL, DMODEL, 256,
                    (long long)NTOK * DMODEL, 0, 0, 0, (long long)NTOK * 256, 0, 1, NBATCH,
                    0, 1.f, 2, nullptr, pb1 + i * 256, nullptr, 0, 0, 0);
        launch_gemm(stream, zz1, pw2 + (size_t)i * 128 * 256, zz2,
                    NTOK, 128, 256, 256, 256, 128,
                    (long long)NTOK * 256, 0, 0, 0, (long long)NTOK * 128, 0, 1, NBATCH,
                    0, 1.f, 2, nullptr, pb2 + i * 128, nullptr, 0, 0, 0);
        decision_kernel<<<(NBATCH * NTOK + 255) / 256, 256, 0, stream>>>(
            zz2, pw3 + (size_t)i * 2 * 128, pb3 + (size_t)i * 2, prevb, polb);

        // ---- attention
        ln_kernel<<<NBATCH * PTOK, 256, 0, stream>>>(Hb, tmpA,
            ln1_g + i * DMODEL, ln1_b + i * DMODEL, PTOK, HS, HS, DMODEL);
        launch_gemm(stream, tmpA, wq + (size_t)i * 256 * DMODEL, qb,
                    PTOK, 256, DMODEL, DMODEL, DMODEL, 256,
                    HS, 0, 0, 0, QS, 0, 1, NBATCH,
                    0, 1.f, 0, nullptr, nullptr, nullptr, 0, 0, 0);
        launch_gemm(stream, tmpA, wk + (size_t)i * 256 * DMODEL, kb,
                    PTOK, 256, DMODEL, DMODEL, DMODEL, 256,
                    HS, 0, 0, 0, QS, 0, 1, NBATCH,
                    0, 1.f, 0, nullptr, nullptr, nullptr, 0, 0, 0);
        launch_gemm(stream, tmpA, wv + (size_t)i * DMODEL * DMODEL, vb,
                    PTOK, DMODEL, DMODEL, DMODEL, DMODEL, DMODEL,
                    HS, 0, 0, 0, HS, 0, 1, NBATCH,
                    0, 1.f, 0, nullptr, nullptr, nullptr, 0, 0, 0);
        // att = (q @ k^T) / sqrt(DK), batched over z = b*H + h
        launch_gemm(stream, qb, kb, attb,
                    PTOK, PTOK, DKH, 256, 256, PPAD,
                    QS, DKH, QS, DKH,
                    (long long)NHEAD * AS, AS, NHEAD, NBATCH * NHEAD,
                    0, 0.17677669529663689f, 0, nullptr, nullptr, nullptr, 0, 0, 0);
        softmax_policy_kernel<<<NBATCH * NHEAD * PTOK, 256, 0, stream>>>(attb, polb);
        // o[b,l,h*64+v] = att @ V; K runs over PPAD (pad cols zeroed by softmax)
        launch_gemm(stream, attb, vb, ob,
                    PTOK, DVH, PPAD, PPAD, DMODEL, DMODEL,
                    (long long)NHEAD * AS, AS,
                    HS, DVH, HS, DVH, NHEAD, NBATCH * NHEAD,
                    1, 1.f, 0, nullptr, nullptr, nullptr, 0, 0, 0);
        // h = o @ wfc^T + h
        launch_gemm(stream, ob, wfc + (size_t)i * DMODEL * DMODEL, Hb,
                    PTOK, DMODEL, DMODEL, DMODEL, DMODEL, DMODEL,
                    HS, 0, 0, 0, HS, 0, 1, NBATCH,
                    0, 1.f, 3, nullptr, nullptr, Hb, HS, 0, DMODEL);

        // ---- FFN
        ln_kernel<<<NBATCH * PTOK, 256, 0, stream>>>(Hb, tmpA,
            ln2_g + i * DMODEL, ln2_b + i * DMODEL, PTOK, HS, HS, DMODEL);
        launch_gemm(stream, tmpA, ffw1 + (size_t)i * DFFN * DMODEL, ff,
                    PTOK, DFFN, DMODEL, DMODEL, DMODEL, DFFN,
                    HS, 0, 0, 0, FS, 0, 1, NBATCH,
                    0, 1.f, 2, nullptr, ffb1 + i * DFFN, nullptr, 0, 0, 0);
        float*    Cout = (i == 3) ? (float*)d_out : Hb;
        long long cs1  = (i == 3) ? (long long)PTOK * DMODEL : HS;
        launch_gemm(stream, ff, ffw2 + (size_t)i * DMODEL * DFFN, Cout,
                    PTOK, DMODEL, DFFN, DFFN, DFFN, DMODEL,
                    FS, 0, 0, 0, cs1, 0, 1, NBATCH,
                    0, 1.f, 3, nullptr, ffb2 + i * DMODEL, Hb, HS, 0, DMODEL);
    }
}
